// EventSampler_53549652246996
// MI455X (gfx1250) — compile-verified, matrix operand order checked by reading, not run on hardware
//
#include <hip/hip_runtime.h>
#include <math.h>

typedef __attribute__((ext_vector_type(16))) _Float16 v16h;
typedef __attribute__((ext_vector_type(8)))  float    v8f;

constexpr int kB = 16;
constexpr int kL = 128;
constexpr int kM = 32;
constexpr int kS = 32;    // NUM_SAMPLE
constexpr int kE = 256;   // NUM_EXP
constexpr int kNB = 20;   // NUM_BOUND
constexpr float kOSR = 1.5f;

__device__ __forceinline__ float softplus_f(float x) {
    // Stable softplus: max(x,0) + log(1 + exp(-|x|)).
    // exp(-|x|) is in (0,1], so log(1+v) has no cancellation -> fast v_log_f32
    // (avoids the libm log1pf polynomial expansion).
    return fmaxf(x, 0.0f) + __logf(1.0f + __expf(-fabsf(x)));
}

__device__ __forceinline__ float fast_rcp(float x) {
    // Bare v_rcp_f32 (1 ulp-ish) -- __fdividef/__frcp_rn still expand to the
    // correctly-rounded div_scale/Newton sequence on this toolchain.
    return __builtin_amdgcn_rcpf(x);
}

// One wave (32 lanes) handles one (b,l) row. All intermediates live in LDS.
__global__ __launch_bounds__(32)
void EventSampler_53549652246996_kernel(
    const float* __restrict__ time_delta,     // [B*L]
    const int*   __restrict__ event_seq,      // [B*L]
    const float* __restrict__ dtime_boundary, // [B*L]
    const float* __restrict__ exp_raw,        // [B*L, E]
    const float* __restrict__ unif,           // [B*L, S, E]
    const float* __restrict__ mu,             // [M]
    const float* __restrict__ alpha,          // [M]
    const float* __restrict__ beta,           // [M]
    const float* __restrict__ type_emb,       // [M, M]
    float* __restrict__ out_res,              // [B*L, S]
    float* __restrict__ out_w)                // [B*L, S]
{
    __shared__ float s_mu[kM], s_cf[kM], s_be[kM];  // cf = alpha*emb
    __shared__ float s_scale[kE];                   // bound / total_intensity[e]
    __shared__ float s_exp[kE];                     // exp_numbers[e]
    __shared__ float s_red[32];

    const int row  = blockIdx.x;       // b*L + l
    const int lane = threadIdx.x;      // 0..31

    // Prefetch this lane's 1KB unif row into the cache hierarchy early.
    const float* urow = unif + (size_t)row * kS * kE + (size_t)lane * kE;
    __builtin_prefetch(urow, 0, 0);

    // --- per-row parameters into LDS (gather emb row via event type) ---
    const int ev = event_seq[row];
    s_mu[lane] = mu[lane];
    s_be[lane] = beta[lane];
    s_cf[lane] = alpha[lane] * type_emb[ev * kM + lane];
    __syncthreads();

    const float td = time_delta[row];

    // --- intensity upper bound: max over 20 linspace points of sum_m softplus ---
    float lsum = -1e30f;
    if (lane < kNB) {
        const float dt = td * ((float)lane * (1.0f / (kNB - 1)));
        float acc = 0.0f;
        #pragma unroll
        for (int m = 0; m < kM; ++m) {
            acc += softplus_f(s_mu[m] + s_cf[m] * __expf(-s_be[m] * dt));
        }
        lsum = acc;
    }
    s_red[lane] = lsum;
    __syncthreads();
    float bmax = s_red[0];
    #pragma unroll
    for (int i = 1; i < kNB; ++i) bmax = fmaxf(bmax, s_red[i]);
    const float bound = bmax * kOSR;
    const float inv_bound = fast_rcp(bound);

    // --- exp_numbers = exp_raw / bound ---
    const float* eraw = exp_raw + (size_t)row * kE;
    #pragma unroll
    for (int k = 0; k < kE / 32; ++k) {
        const int e = lane + 32 * k;
        s_exp[e] = eraw[e] * inv_bound;
    }
    __syncthreads();

    // --- total intensities via WMMA ones-reduction over M ---
    // Tile t covers exp-samples e = t*16 .. t*16+15 (A rows). Lane i computes
    // m=0..15 of row i, lane i+16 computes m=16..31 of row i. With B == ones,
    // D[i][j] = sum_k A[i][k] regardless of slot->K mapping, so per-lane slot
    // order is irrelevant. D cols are identical; lane 0 holds rows 0..7 in its
    // 8 accumulator VGPRs, lane 16 holds rows 8..15.
    const int mbase = (lane >= 16) ? 16 : 0;
    float r_mu[16], r_cf[16], r_be[16];
    #pragma unroll
    for (int i = 0; i < 16; ++i) {
        r_mu[i] = s_mu[mbase + i];
        r_cf[i] = s_cf[mbase + i];
        r_be[i] = s_be[mbase + i];
    }
    v16h ones;
    #pragma unroll
    for (int i = 0; i < 16; ++i) ones[i] = (_Float16)1.0f;

    const int eInTile = lane & 15;
    for (int t = 0; t < kE / 16; ++t) {
        const float dt = s_exp[t * 16 + eInTile];
        v16h a;
        #pragma unroll
        for (int i = 0; i < 16; ++i) {
            a[i] = (_Float16)softplus_f(r_mu[i] + r_cf[i] * __expf(-r_be[i] * dt));
        }
        v8f c = {};
        c = __builtin_amdgcn_wmma_f32_16x16x32_f16(
                /*neg_a=*/false, a, /*neg_b=*/false, ones,
                /*c_mod=*/(short)0, c, /*reuse_a=*/false, /*reuse_b=*/false);
        if (lane == 0 || lane == 16) {
            const int base = t * 16 + ((lane >= 16) ? 8 : 0);
            #pragma unroll
            for (int i = 0; i < 8; ++i) s_scale[base + i] = c[i];
        }
    }
    __syncthreads();

    // Replace totals with bound/total so the hot loop multiplies, not divides.
    // Bare v_rcp_f32 + mul: no div_scale/Newton chain.
    #pragma unroll
    for (int k = 0; k < kE / 32; ++k) {
        const int e = lane + 32 * k;
        s_scale[e] = bound * fast_rcp(s_scale[e]);
    }
    __syncthreads();

    // --- accept/reject: lane == sample s; scan 256 candidate times ---
    // criterion[s,e] = unif[s,e] * bound / total[e]; accepted iff crit < 1.
    // The reference's `big` sentinel never reaches the output (rows with no
    // acceptance route to dtb/last_exp), so only the min over accepted
    // exp_numbers is needed.
    float accMin = 1e30f;
    bool  any = false;
    for (int e = 0; e < kE; ++e) {
        const float crit = urow[e] * s_scale[e];
        if (crit < 1.0f) {
            any = true;
            accMin = fminf(accMin, s_exp[e]);
        }
    }
    const float dtb  = dtime_boundary[row];
    const float last = s_exp[kE - 1];
    float res = any ? accMin : ((last > dtb) ? last : dtb);
    res = fminf(res, 1e5f);

    out_res[(size_t)row * kS + lane] = res;
    out_w  [(size_t)row * kS + lane] = 1.0f / kS;
}

extern "C" void kernel_launch(void* const* d_in, const int* in_sizes, int n_in,
                              void* d_out, int out_size, void* d_ws, size_t ws_size,
                              hipStream_t stream) {
    // setup_inputs() order:
    // 0: time_seq (unused), 1: time_delta_seq, 2: event_seq(int), 3: dtime_boundary,
    // 4: exp_raw, 5: unif_numbers, 6: mu, 7: alpha, 8: beta, 9: type_emb
    const float* time_delta = (const float*)d_in[1];
    const int*   event_seq  = (const int*)  d_in[2];
    const float* dtb        = (const float*)d_in[3];
    const float* exp_raw    = (const float*)d_in[4];
    const float* unif       = (const float*)d_in[5];
    const float* mu         = (const float*)d_in[6];
    const float* alpha      = (const float*)d_in[7];
    const float* beta       = (const float*)d_in[8];
    const float* type_emb   = (const float*)d_in[9];

    float* out_res = (float*)d_out;                      // [B,L,S]
    float* out_w   = out_res + (size_t)kB * kL * kS;     // [B,L,S]

    dim3 grid(kB * kL);   // 2048 rows, one wave32 each
    dim3 block(32);
    EventSampler_53549652246996_kernel<<<grid, block, 0, stream>>>(
        time_delta, event_seq, dtb, exp_raw, unif,
        mu, alpha, beta, type_emb, out_res, out_w);
}
